// TransformerDecoder_59768764891289
// MI455X (gfx1250) — matrix operand order repaired and checked by reading, not compiled, not run on hardware
//
#include <hip/hip_runtime.h>
#include <hip/hip_bf16.h>
#include <stdint.h>

// ---------------------------------------------------------------------------
// MI455X (gfx1250, wave32) point-cloud decoder.
//  - All GEMMs via v_wmma_f32_16x16x32_bf16, fp32 accumulation.
//  - Wave tile 32x64 (2x4 fragments -> 8 WMMAs / K-step, 1.5 LDS b128 loads
//    per WMMA); 8 waves (4Mx2N) tile 128x128 per workgroup.
//  - GEMM operands bf16 in memory; activation rows padded to NBP=6912 so
//    16B-aligned global_load_async_to_lds_b128 (ASYNCcnt) staging is legal.
//  - Double-buffered: s_wait_asynccnt 0 -> barrier -> issue next async copies
//    -> ds_load_b128 fragments + 8 WMMAs -> barrier.
// ---------------------------------------------------------------------------

typedef __attribute__((ext_vector_type(16))) __bf16 v16bf;
typedef __attribute__((ext_vector_type(8)))  __bf16 v8bf;
typedef __attribute__((ext_vector_type(8)))  float  v8f;

#define NPTS 6890
#define NBP  6912     // bf16 row stride (16B aligned: 6912*2 = 13824 = 864*16)
#define EPS  1e-5f

#define RS_A 40       // LDS row stride (elems) for A tile [128][32]
#define RS_B 40       // LDS row stride (elems) for B tile [128][32] ([col][k])

enum { EP_NONE = 0, EP_BIAS = 1, EP_BIAS_RES = 2, EP_GAMMA_RES = 3, EP_TANH2 = 4 };
enum { OUT_F32 = 0, OUT_BF16 = 1, OUT_BOTH = 2 };

// ---- CDNA5 async global->LDS copy (ASYNCcnt) ------------------------------
__device__ __forceinline__ void async_copy_b128(uint32_t lds_addr, const void* g)
{
    asm volatile("global_load_async_to_lds_b128 %0, %1, off"
                 :: "v"(lds_addr), "v"(g) : "memory");
}
__device__ __forceinline__ void wait_async0()
{
    asm volatile("s_wait_asynccnt 0" ::: "memory");
}
// generic LDS pointer -> 32-bit LDS offset (flat LDS aperture: addr[31:0])
__device__ __forceinline__ uint32_t lds_off_u32(const void* p)
{
    return (uint32_t)(uintptr_t)p;
}

// ----------------------------- WMMA GEMM -----------------------------------
// C[row,col] = sum_k A(row,k)*B(k,col) (+ epilogue).  A/B bf16, C f32/bf16.
__global__ __launch_bounds__(256)
void wmma_gemm_kernel(const __bf16* __restrict__ A, long a_rs, long a_cs,
                      const __bf16* __restrict__ B, long b_rs, long b_cs,
                      float* __restrict__ Cf, long cf_rs,
                      __bf16* __restrict__ Cb, long cb_rs,
                      int M, int N, int K, int mode, int outkind,
                      const float* __restrict__ bias,
                      const float* __restrict__ res, long res_rs,
                      const float* __restrict__ gamma)
{
    __shared__ __bf16 As[2][128 * RS_A];
    __shared__ __bf16 Bs[2][128 * RS_B];

    const int tid   = threadIdx.x;
    const int lane  = tid & 31;
    const int wid   = tid >> 5;
    const int wm    = wid & 3;           // 4 waves along M (32 rows each)
    const int wn    = wid >> 2;          // 2 waves along N (64 cols each)
    const int m0b   = (int)blockIdx.x * 128;
    const int n0b   = (int)blockIdx.y * 128;
    const int lrow  = lane & 15;
    const int lhalf = lane >> 4;

    const bool edgeA_mn = (m0b + 128 > M);
    const bool edgeB_mn = (n0b + 128 > N);
    const uint32_t asBase = lds_off_u32(&As[0][0]);
    const uint32_t bsBase = lds_off_u32(&Bs[0][0]);
    constexpr int AS_BUF = 128 * RS_A;
    constexpr int BS_BUF = 128 * RS_B;

    auto stageA = [&](int buf, int k0, bool guard) {
        if (!guard && a_cs == 1) {
            // async: 512 16B-chunks (128 rows x 4 k-chunks), 2 per thread
            #pragma unroll
            for (int i = 0; i < 2; ++i) {
                const int idx = i * 256 + tid;
                const int row = idx >> 2, c = idx & 3;
                const __bf16* g = A + (long)(m0b + row) * a_rs + (k0 + c * 8);
                async_copy_b128(asBase + (uint32_t)(buf * AS_BUF + row * RS_A + c * 8) * 2u, g);
            }
        } else if (!guard) {
            // a_rs==1: vectorized transpose staging (contiguous along m)
            #pragma unroll
            for (int i = 0; i < 2; ++i) {
                const int idx = i * 256 + tid;
                const int kk = idx >> 4, rr = idx & 15;
                v8bf d = *(const v8bf*)(A + (long)(m0b + rr * 8) + (long)(k0 + kk) * a_cs);
                #pragma unroll
                for (int j = 0; j < 8; ++j) As[buf][(rr * 8 + j) * RS_A + kk] = d[j];
            }
        } else {
            #pragma unroll
            for (int i = 0; i < 16; ++i) {
                const int e = i * 256 + tid;
                int row, kk;
                if (a_cs == 1) { row = e >> 5;  kk = e & 31; }
                else           { row = e & 127; kk = e >> 7; }
                const int gm = m0b + row, gk = k0 + kk;
                __bf16 val = (__bf16)0.0f;
                if (gm < M && gk < K) val = A[(long)gm * a_rs + (long)gk * a_cs];
                As[buf][row * RS_A + kk] = val;
            }
        }
    };
    auto stageB = [&](int buf, int k0, bool guard) {
        if (!guard && b_rs == 1) {
            // async: 512 16B-chunks (128 cols x 4 k-chunks), 2 per thread
            #pragma unroll
            for (int i = 0; i < 2; ++i) {
                const int idx = i * 256 + tid;
                const int col = idx >> 2, c = idx & 3;
                const __bf16* g = B + (long)(k0 + c * 8) + (long)(n0b + col) * b_cs;
                async_copy_b128(bsBase + (uint32_t)(buf * BS_BUF + col * RS_B + c * 8) * 2u, g);
            }
        } else if (!guard) {
            // b_cs==1: vectorized transpose staging (contiguous along n)
            #pragma unroll
            for (int i = 0; i < 2; ++i) {
                const int idx = i * 256 + tid;
                const int kk = idx >> 4, cc = idx & 15;
                v8bf d = *(const v8bf*)(B + (long)(k0 + kk) * b_rs + (n0b + cc * 8));
                #pragma unroll
                for (int j = 0; j < 8; ++j) Bs[buf][(cc * 8 + j) * RS_B + kk] = d[j];
            }
        } else {
            #pragma unroll
            for (int i = 0; i < 16; ++i) {
                const int e = i * 256 + tid;
                int col, kk;
                if (b_cs == 1) { col = e & 127; kk = e >> 7; }
                else           { col = e >> 5;  kk = e & 31; }
                const int gn = n0b + col, gk = k0 + kk;
                __bf16 val = (__bf16)0.0f;
                if (gn < N && gk < K) val = B[(long)gk * b_rs + (long)gn * b_cs];
                Bs[buf][col * RS_B + kk] = val;
            }
        }
    };

    // fragment loads: two ds_load_b128 each (ISA 7.12.2 layouts)
    auto ldfragA = [&](int buf, int t) -> v16bf {
        const __bf16* p = &As[buf][(wm * 32 + t * 16 + lrow) * RS_A + lhalf * 8];
        v8bf lo = *(const v8bf*)p;
        v8bf hi = *(const v8bf*)(p + 16);
        v16bf f;
        #pragma unroll
        for (int i = 0; i < 8; ++i) { f[i] = lo[i]; f[8 + i] = hi[i]; }
        return f;
    };
    auto ldfragB = [&](int buf, int t) -> v16bf {
        const __bf16* p = &Bs[buf][(wn * 64 + t * 16 + lrow) * RS_B + lhalf * 8];
        v8bf lo = *(const v8bf*)p;
        v8bf hi = *(const v8bf*)(p + 16);
        v16bf f;
        #pragma unroll
        for (int i = 0; i < 8; ++i) { f[i] = lo[i]; f[8 + i] = hi[i]; }
        return f;
    };

    v8f acc[2][4] = {};
    const int nk = (K + 31) >> 5;
    stageA(0, 0, edgeA_mn || (32 > K));
    stageB(0, 0, edgeB_mn || (32 > K));

    int cur = 0;
    for (int kt = 0; kt < nk; ++kt) {
        wait_async0();          // current buffer's async copies done
        __syncthreads();        // visible to all waves (dscnt waited by compiler)

        if (kt + 1 < nk) {      // overlap next tile's copies with the WMMAs
            const int kn = (kt + 1) << 5;
            stageA(cur ^ 1, kn, edgeA_mn || (kn + 32 > K));
            stageB(cur ^ 1, kn, edgeB_mn || (kn + 32 > K));
        }

        v16bf a0 = ldfragA(cur, 0);
        v16bf a1 = ldfragA(cur, 1);
        v16bf bfr[4];
        #pragma unroll
        for (int j = 0; j < 4; ++j) bfr[j] = ldfragB(cur, j);

        #pragma unroll
        for (int j = 0; j < 4; ++j) {
            acc[0][j] = __builtin_amdgcn_wmma_f32_16x16x32_bf16(
                false, a0, false, bfr[j], (short)0, acc[0][j], false, false);
            acc[1][j] = __builtin_amdgcn_wmma_f32_16x16x32_bf16(
                false, a1, false, bfr[j], (short)0, acc[1][j], false, false);
        }

        __syncthreads();        // all reads of cur done before it is re-staged
        cur ^= 1;
    }

    const float g = (mode == EP_GAMMA_RES) ? gamma[0] : 0.0f;
    #pragma unroll
    for (int i = 0; i < 2; ++i) {
        #pragma unroll
        for (int j = 0; j < 4; ++j) {
            const int col = n0b + wn * 64 + j * 16 + lrow;
            #pragma unroll
            for (int r = 0; r < 8; ++r) {
                const int row = m0b + wm * 32 + i * 16 + lhalf * 8 + r;
                if (row < M && col < N) {
                    float v = acc[i][j][r];
                    switch (mode) {
                        case EP_BIAS:      v += bias[row]; break;
                        case EP_BIAS_RES:  v += bias[row] + res[(long)row * res_rs + col]; break;
                        case EP_GAMMA_RES: v = res[(long)row * res_rs + col] + g * v; break;
                        case EP_TANH2:     v = 2.0f * tanhf(v + bias[row]); break;
                        default: break;
                    }
                    if (outkind != OUT_BF16) Cf[(long)row * cf_rs + col] = v;
                    if (outkind != OUT_F32)  Cb[(long)row * cb_rs + col] = (__bf16)v;
                }
            }
        }
    }
}

// ------------------------ f32 -> bf16 row convert --------------------------
__global__ __launch_bounds__(256)
void f32_to_bf16_rows(const float* __restrict__ in, long in_rs,
                      __bf16* __restrict__ out, long out_rs, int cols)
{
    const long r = blockIdx.x;
    const float* s = in + r * in_rs;
    __bf16* d = out + r * out_rs;
    for (int i = threadIdx.x; i < cols; i += 256) d[i] = (__bf16)s[i];
}

// --------------------------- block reductions ------------------------------
__device__ __forceinline__ float block_reduce_sum(float v, float* sm)
{
    #pragma unroll
    for (int off = 16; off > 0; off >>= 1) v += __shfl_xor(v, off, 32);
    const int wid = threadIdx.x >> 5, lane = threadIdx.x & 31;
    if (lane == 0) sm[wid] = v;
    __syncthreads();
    if (wid == 0) {
        float t = (lane < 8) ? sm[lane] : 0.0f;
        #pragma unroll
        for (int off = 4; off > 0; off >>= 1) t += __shfl_xor(t, off, 32);
        if (lane == 0) sm[0] = t;
    }
    __syncthreads();
    float r = sm[0];
    __syncthreads();
    return r;
}
__device__ __forceinline__ float block_reduce_max(float v, float* sm)
{
    #pragma unroll
    for (int off = 16; off > 0; off >>= 1) v = fmaxf(v, __shfl_xor(v, off, 32));
    const int wid = threadIdx.x >> 5, lane = threadIdx.x & 31;
    if (lane == 0) sm[wid] = v;
    __syncthreads();
    if (wid == 0) {
        float t = (lane < 8) ? sm[lane] : -3.402823466e38f;
        #pragma unroll
        for (int off = 4; off > 0; off >>= 1) t = fmaxf(t, __shfl_xor(t, off, 32));
        if (lane == 0) sm[0] = t;
    }
    __syncthreads();
    float r = sm[0];
    __syncthreads();
    return r;
}

// ------------------------------ softmax ------------------------------------
// row softmax of f32 E -> bf16 Eb (padded row stride)
__global__ __launch_bounds__(256)
void softmax_rows_kernel(const float* __restrict__ E, __bf16* __restrict__ Eb,
                         int n, long orow)
{
    __shared__ float sm[8];
    const float* p = E + (long)blockIdx.x * n;
    __bf16* q = Eb + (long)blockIdx.x * orow;
    const int tid = threadIdx.x;

    float m = -3.402823466e38f;
    for (int i = tid; i < n; i += 256) m = fmaxf(m, p[i]);
    m = block_reduce_max(m, sm);

    float s = 0.0f;
    for (int i = tid; i < n; i += 256) {
        float e = __expf(p[i] - m);
        q[i] = (__bf16)e;
        s += e;
    }
    s = block_reduce_sum(s, sm);
    const float inv = 1.0f / s;
    for (int i = tid; i < n; i += 256) q[i] = (__bf16)((float)q[i] * inv);
}

// ----------------- fused InstanceNorm + SPADAIN + ReLU ---------------------
// f32 in (stride n), bf16 out (stride orow)
__global__ __launch_bounds__(256)
void inorm_spadain_relu_kernel(const float* __restrict__ X,
                               const float* __restrict__ add,
                               const float* __restrict__ ww, const float* __restrict__ wb,
                               const float* __restrict__ bw, const float* __restrict__ bb,
                               __bf16* __restrict__ Y, int n, long orow)
{
    __shared__ float sm[8];
    const int c = blockIdx.x;
    const float* x = X + (long)c * n;
    __bf16* y = Y + (long)c * orow;
    const int tid = threadIdx.x;

    float s = 0.0f, sq = 0.0f;
    for (int i = tid; i < n; i += 256) { float v = x[i]; s += v; sq += v * v; }
    s  = block_reduce_sum(s, sm);
    sq = block_reduce_sum(sq, sm);
    const float mean = s / (float)n;
    const float var  = sq / (float)n - mean * mean;
    const float inv  = rsqrtf(var + EPS);

    const float w0 = ww[c * 3 + 0], w1 = ww[c * 3 + 1], w2 = ww[c * 3 + 2], wc = wb[c];
    const float b0 = bw[c * 3 + 0], b1 = bw[c * 3 + 1], b2 = bw[c * 3 + 2], bc = bb[c];
    const float* a0 = add;
    const float* a1 = add + n;
    const float* a2 = add + 2 * (long)n;

    for (int i = tid; i < n; i += 256) {
        const float p0 = a0[i], p1 = a1[i], p2 = a2[i];
        const float ws = w0 * p0 + w1 * p1 + w2 * p2 + wc;
        const float bs = b0 * p0 + b1 * p1 + b2 * p2 + bc;
        float v = ws * ((x[i] - mean) * inv) + bs;
        y[i] = (__bf16)fmaxf(v, 0.0f);
    }
}

// ------------------------------- host side ---------------------------------
struct AttP { const float *gamma, *kb, *kw, *qb, *qw, *vb, *vw; };
struct SpP  { const float *bb, *bw, *wb, *ww; };
struct ResP { const float *c1b, *c1w, *c2b, *c2w, *crb, *crw; SpP s1, s2, sr; };

extern "C" void kernel_launch(void* const* d_in, const int* in_sizes, int n_in,
                              void* d_out, int out_size, void* d_ws, size_t ws_size,
                              hipStream_t stream)
{
    (void)in_sizes; (void)n_in; (void)out_size; (void)ws_size;
    auto F = [&](int i) { return (const float*)d_in[i]; };

    // jax tree order (sorted dict keys): 0: addition, 1..110: params, 111: x1_f
    const float* addition = F(0);
    int ix = 1;
    AttP att[4];
    for (int a = 0; a < 4; ++a) {
        att[a].gamma = F(ix++); att[a].kb = F(ix++); att[a].kw = F(ix++);
        att[a].qb = F(ix++);    att[a].qw = F(ix++);
        att[a].vb = F(ix++);    att[a].vw = F(ix++);
    }
    const float *conv_b[5], *conv_w[5];
    for (int cidx = 0; cidx < 5; ++cidx) { conv_b[cidx] = F(ix++); conv_w[cidx] = F(ix++); }
    ResP sp[4];
    for (int r = 0; r < 4; ++r) {
        sp[r].c1b = F(ix++); sp[r].c1w = F(ix++);
        sp[r].c2b = F(ix++); sp[r].c2w = F(ix++);
        sp[r].crb = F(ix++); sp[r].crw = F(ix++);
        SpP* sub[3] = { &sp[r].s1, &sp[r].s2, &sp[r].sr };
        for (int s = 0; s < 3; ++s) {
            sub[s]->bb = F(ix++); sub[s]->bw = F(ix++);
            sub[s]->wb = F(ix++); sub[s]->ww = F(ix++);
        }
    }
    const float* x1_f = F(ix);

    // ---- workspace layout ----
    float* E   = (float*)d_ws;
    float* xf  = E   + (size_t)NPTS * NPTS;
    float* af  = xf  + (size_t)1024 * NPTS;
    float* tf1 = af  + (size_t)1024 * NPTS;
    float* tf2 = tf1 + (size_t)1024 * NPTS;
    __bf16* Eb  = (__bf16*)(tf2 + (size_t)1024 * NPTS);
    __bf16* rb  = Eb  + (size_t)NPTS * NBP;
    __bf16* xb  = rb  + (size_t)1024 * NBP;
    __bf16* qb  = xb  + (size_t)1024 * NBP;
    __bf16* kb  = qb  + (size_t)1024 * NBP;
    __bf16* vb  = kb  + (size_t)1024 * NBP;
    __bf16* sb1 = vb  + (size_t)1024 * NBP;
    __bf16* sb3 = sb1 + (size_t)1024 * NBP;
    __bf16* wscr = sb3 + (size_t)1024 * NBP;

    auto gemm = [&](const __bf16* A, long ars, long acs,
                    const __bf16* B, long brs, long bcs,
                    float* Cf, long cfrs, __bf16* Cb, long cbrs,
                    int M, int N, int K, int mode, int outk,
                    const float* bias, const float* res, long res_rs,
                    const float* gamma) {
        dim3 grid((M + 127) / 128, (N + 127) / 128), block(256);
        wmma_gemm_kernel<<<grid, block, 0, stream>>>(A, ars, acs, B, brs, bcs,
                                                     Cf, cfrs, Cb, cbrs,
                                                     M, N, K, mode, outk,
                                                     bias, res, res_rs, gamma);
    };
    auto cvtW = [&](const float* w, int rows, int cols) {
        f32_to_bf16_rows<<<dim3(rows), dim3(256), 0, stream>>>(w, cols, wscr, cols, cols);
    };
    auto conv = [&](const float* Wf, const float* bvec, const __bf16* Xb,
                    float* Yf, __bf16* Yb, int Co, int Ci, int outk) {
        cvtW(Wf, Co, Ci);
        gemm(wscr, Ci, 1, Xb, NBP, 1, Yf, NPTS, Yb, NBP, Co, NPTS, Ci,
             EP_BIAS, outk, bvec, nullptr, 0, nullptr);
    };
    auto attention = [&](const float* Xfp, const __bf16* Xbp, float* Yf,
                         const AttP& ap, int C) {
        conv(ap.qw, ap.qb, Xbp, nullptr, qb, C, C, OUT_BF16);
        conv(ap.kw, ap.kb, Xbp, nullptr, kb, C, C, OUT_BF16);
        conv(ap.vw, ap.vb, Xbp, nullptr, vb, C, C, OUT_BF16);
        // energy E(n,m) = sum_c q[c,n]*k[c,m]; A=q^T (1, NBP), B=k (NBP, 1)
        gemm(qb, 1, NBP, kb, NBP, 1, E, NPTS, nullptr, 0, NPTS, NPTS, C,
             EP_NONE, OUT_F32, nullptr, nullptr, 0, nullptr);
        softmax_rows_kernel<<<dim3(NPTS), dim3(256), 0, stream>>>(E, Eb, NPTS, NBP);
        // out[c,m] = Xf[c,m] + gamma * sum_n v[c,n]*att(m,n); att(m,n)=Eb[m*NBP+n]
        gemm(vb, NBP, 1, Eb, 1, NBP, Yf, NPTS, nullptr, 0, C, NPTS, NPTS,
             EP_GAMMA_RES, OUT_F32, nullptr, Xfp, NPTS, ap.gamma);
    };
    auto spadain = [&](const float* Xfp, const SpP& s, __bf16* Yb, int C) {
        inorm_spadain_relu_kernel<<<dim3(C), dim3(256), 0, stream>>>(
            Xfp, addition, s.ww, s.wb, s.bw, s.bb, Yb, NPTS, NBP);
    };
    auto resblock = [&](const float* Xfp, const ResP& rp, __bf16* Yb, int C) {
        spadain(Xfp, rp.s1, sb1, C);
        cvtW(rp.c1w, C, C);
        gemm(wscr, C, 1, sb1, NBP, 1, tf1, NPTS, nullptr, 0, C, NPTS, C,
             EP_BIAS, OUT_F32, rp.c1b, nullptr, 0, nullptr);
        spadain(tf1, rp.s2, sb1, C);
        spadain(Xfp, rp.sr, sb3, C);
        cvtW(rp.crw, C, C);
        gemm(wscr, C, 1, sb3, NBP, 1, tf2, NPTS, nullptr, 0, C, NPTS, C,
             EP_BIAS, OUT_F32, rp.crb, nullptr, 0, nullptr);
        cvtW(rp.c2w, C, C);
        gemm(wscr, C, 1, sb1, NBP, 1, nullptr, 0, Yb, NBP, C, NPTS, C,
             EP_BIAS_RES, OUT_BF16, rp.c2b, tf2, NPTS, nullptr);
    };

    // ------------------------- decoder pipeline ---------------------------
    f32_to_bf16_rows<<<dim3(1024), dim3(256), 0, stream>>>(x1_f, NPTS, rb, NBP, NPTS);

    conv(conv_w[0], conv_b[0], rb, xf, xb, 1024, 1024, OUT_BOTH);  // conv1
    attention(xf, xb, af, att[0], 1024);                           // att1 -> af
    resblock(af, sp[0], rb, 1024);                                 // sp1  -> rb

    conv(conv_w[1], conv_b[1], rb, xf, xb, 512, 1024, OUT_BOTH);   // conv2
    attention(xf, xb, af, att[1], 512);
    resblock(af, sp[1], rb, 512);

    conv(conv_w[2], conv_b[2], rb, xf, xb, 512, 512, OUT_BOTH);    // conv3
    attention(xf, xb, af, att[2], 512);
    resblock(af, sp[2], rb, 512);

    conv(conv_w[3], conv_b[3], rb, xf, xb, 256, 512, OUT_BOTH);    // conv4
    attention(xf, xb, af, att[3], 256);
    resblock(af, sp[3], rb, 256);

    // conv5 + 2*tanh -> d_out [3, NPTS]
    cvtW(conv_w[4], 3, 256);
    gemm(wscr, 256, 1, rb, NBP, 1, (float*)d_out, NPTS, nullptr, 0,
         3, NPTS, 256, EP_TANH2, OUT_F32, conv_b[4], nullptr, 0, nullptr);
}
